// Model_52931176955977
// MI455X (gfx1250) — compile-verified, matrix operand order checked by reading, not run on hardware
//
#include <hip/hip_runtime.h>

// ============================================================================
// Fused fixed-point LSTM (B=4096, T=2048, I=3, H=5) for MI455X (gfx1250).
//
//  * Mandatory HBM traffic: read x (100.7 MB) + write y (33.6 MB) ~ 134 MB
//    -> ~5.8 us at 23.3 TB/s. Weights are loop-invariant and tiny.
//  * The T=2048 recurrence is serial per batch row (4096 rows = 128 wave32's)
//    so wall time = per-step dependence-chain latency x 2048. Everything is
//    fused (no 671 MB x_proj tensor); input projection + recurrent matmul
//    fold into ONE v_wmma_f32_16x16x32_f16 per 16x16 tile with K = I+H = 8:
//        16*gates[b][j] = [xq(t) | 16*h(t-1)] @ [16*W_ih^T ; W_hh^T] + 16*bias
//    All operands are exact in f16 (grid values / small integers); WMMA
//    multiplies into an f32 accumulator -> exact, bit-identical math.
//  * SCALED INTEGER DOMAIN: cell state, hidden state and activations are
//    carried as 16x integers (c16, hn16, ia16/fa16/oa16 in [0,16]); the
//    division-free folded quantizers are:
//      16*q4(hardtanh(q4(g)))    == clamp(rint(16g), -16, 16)
//      16*q4(hardsigmoid(q4(g))) == clamp(rint(rint(16g)/6 + 8), 0, 16)
//      16*c_new = clamp(rint((fa16*c16 + ia16*ga16)/16), -128, 127)
//    (the +-8 fixed-point saturation provably cannot interact with the
//    hard-activation clips on these paths).
//  * A-operand rows live in LDS as 32 B pairs: 8 live f16 K-values + 16 B of
//    permanent zeros (K=8..15), plus an all-zero pad row for lanes 16..31
//    (which carry K=16..31 of the WMMA A tile). Every lane then issues two
//    unconditional 32 B LDS loads from loop-invariant addresses -> no EXEC
//    masking and no per-iteration zero-fill of A registers.
//  * One wave owns 32 batch rows: 2 M-tiles x 2 N-tiles = 4 WMMAs/step with
//    register-resident weights/bias. A 2.5 KB per-wave LDS buffer transposes
//    the WMMA D layout (lane = gate column) to lane = batch row for the cell
//    update. 128 single-wave workgroups -> no barriers; LDS round trips
//    ordered with the gfx1250 split counter s_wait_dscnt.
// ============================================================================

typedef __attribute__((ext_vector_type(16))) _Float16 v16h;
typedef __attribute__((ext_vector_type(8)))  float    v8f;

#define B_TOT 4096
#define T_LEN 2048
#define I_DIM 3
#define H_DIM 5
#define G_DIM 20            // 4*H gate columns
#define K_DIM 8             // I + H packed into one WMMA K slice

// FixedPoint(8,4) quantize (round-half-even like jnp.round, saturate).
__device__ __forceinline__ float q4(float v) {
    float r = rintf(v * 16.0f) * 0.0625f;
    return fminf(fmaxf(r, -8.0f), 7.9375f);
}
// 16*q4(hardsigmoid(q4(.))) given p = rint(16*gate): integer in [0,16].
__device__ __forceinline__ float qhsig16(float p) {
    float t = rintf(__builtin_fmaf(p, 1.0f / 6.0f, 8.0f));
    return fminf(fmaxf(t, 0.0f), 16.0f);
}

struct alignas(32) APair { _Float16 h[16]; };  // live row (K=0..7) + zero pad
struct alignas(16) ARow  { _Float16 h[8];  };  // the live half

__global__ __launch_bounds__(32) void
lstm_qfixed_wmma(const float* __restrict__ x,
                 const float* __restrict__ W_ih,
                 const float* __restrict__ W_hh,
                 const float* __restrict__ b_ih,
                 const float* __restrict__ b_hh,
                 const float* __restrict__ W_lin,
                 const float* __restrict__ b_lin,
                 float* __restrict__ y)
{
    __shared__ float gbuf[32 * G_DIM];   // 32 batch rows x 20 cols of 16*gate
    __shared__ APair hp[34];             // 32 A rows + 2 all-zero pad rows

    const int lane = threadIdx.x;              // 0..31 (wave32)
    const int b    = blockIdx.x * 32 + lane;   // this lane's batch row

    // ---- loop-invariant WMMA operands ------------------------------------
    // B operand (f16 32x16; K=0..7 used): lane l<16 holds gate column
    // j = nt*16 + l;  element k = 16*q(W_ih[j][k]) for k<3 (integer <=128),
    //                 element k = q(W_hh[j][k-3]) for k>=3 (grid value <=8);
    // paired with A = [xq | 16*h] this yields 16*gate. Lanes 16..31 carry
    // K=16..31 -> zero. C operand: 16*(q(b_ih)+q(b_hh)) per gate column
    // (D layout has N = lane&15 in every lane).
    const int jc = lane & 15;
    v16h Bv[2];
    v8f  Cv[2];
#pragma unroll
    for (int nt = 0; nt < 2; ++nt) {
        const int j = nt * 16 + jc;
        v16h bv = {};
        if (lane < 16 && j < G_DIM) {
#pragma unroll
            for (int k = 0; k < I_DIM; ++k)
                bv[k] = (_Float16)(16.0f * q4(W_ih[j * I_DIM + k]));
#pragma unroll
            for (int k = 0; k < H_DIM; ++k)
                bv[I_DIM + k] = (_Float16)q4(W_hh[j * H_DIM + k]);
        }
        Bv[nt] = bv;
        const float bias = (j < G_DIM) ? 16.0f * (q4(b_ih[j]) + q4(b_hh[j])) : 0.0f;
        v8f cv;
#pragma unroll
        for (int r = 0; r < 8; ++r) cv[r] = bias;
        Cv[nt] = cv;
    }

    // Output head (value domain weights, 16x bias; y = clamp(rint(acc16))/16).
    float wl[H_DIM];
#pragma unroll
    for (int k = 0; k < H_DIM; ++k) wl[k] = q4(W_lin[k]);
    const float bl16 = 16.0f * q4(b_lin[0]);

    // ---- LDS A rows: zero everything once (incl. pads), then seed t=0 ----
    {
        APair z;
#pragma unroll
        for (int k = 0; k < 16; ++k) z.h[k] = (_Float16)0.0f;
        hp[lane] = z;
        if (lane < 2) hp[32 + lane] = z;
    }

    const float* xb = x + (size_t)b * T_LEN * I_DIM;
    float*       yb = y + (size_t)b * T_LEN;

    float c16[H_DIM];                    // 16*c, integer-valued floats
#pragma unroll
    for (int k = 0; k < H_DIM; ++k) c16[k] = 0.0f;

    {
        ARow r0;
        r0.h[0] = (_Float16)q4(xb[0]);
        r0.h[1] = (_Float16)q4(xb[1]);
        r0.h[2] = (_Float16)q4(xb[2]);
#pragma unroll
        for (int k = 0; k < H_DIM; ++k) r0.h[I_DIM + k] = (_Float16)0.0f;
        *(ARow*)&hp[lane] = r0;          // h0 = 0
    }

    // Loop-invariant A-gather addresses: lanes 0..15 read rows l and l+16;
    // lanes 16..31 read the permanent zero pads (they hold K=16..31 == 0).
    const v16h* ap0 = (const v16h*)&hp[(lane < 16) ? lane        : 32];
    const v16h* ap1 = (const v16h*)&hp[(lane < 16) ? (lane + 16) : 33];

    const int mrow = (lane >> 4) << 3;   // D layout: row group this lane holds
    const int ncol = lane & 15;          // D layout: gate column this lane holds

    for (int t = 0; t < T_LEN; ++t) {
        // -- gather A rows: two unconditional 32 B LDS loads ----------------
        asm volatile("s_wait_dscnt 0" ::: "memory");
        v16h A0 = *ap0;
        v16h A1 = *ap1;

        // -- 16*gates : four 16x16 WMMA tiles -------------------------------
        v8f d00 = __builtin_amdgcn_wmma_f32_16x16x32_f16(false, A0, false, Bv[0], (short)0, Cv[0], false, false);
        v8f d01 = __builtin_amdgcn_wmma_f32_16x16x32_f16(false, A0, false, Bv[1], (short)0, Cv[1], false, false);
        v8f d10 = __builtin_amdgcn_wmma_f32_16x16x32_f16(false, A1, false, Bv[0], (short)0, Cv[0], false, false);
        v8f d11 = __builtin_amdgcn_wmma_f32_16x16x32_f16(false, A1, false, Bv[1], (short)0, Cv[1], false, false);

        // -- scatter D tiles to LDS so each lane owns one batch row ---------
#pragma unroll
        for (int r = 0; r < 8; ++r) {
            gbuf[(mrow + r) * G_DIM + ncol]      = d00[r];
            gbuf[(16 + mrow + r) * G_DIM + ncol] = d10[r];
        }
        if (ncol < G_DIM - 16) {            // N-tile 1: only columns 16..19
#pragma unroll
            for (int r = 0; r < 8; ++r) {
                gbuf[(mrow + r) * G_DIM + 16 + ncol]      = d01[r];
                gbuf[(16 + mrow + r) * G_DIM + 16 + ncol] = d11[r];
            }
        }
        asm volatile("s_wait_dscnt 0" ::: "memory");

        // -- per-lane fixed-point cell update, 16x integer domain -----------
        float g[G_DIM];                      // g[k] = 16 * gate_raw
        const float* grow = gbuf + lane * G_DIM;   // 80 B stride, 16 B aligned
#pragma unroll
        for (int k = 0; k < G_DIM; ++k) g[k] = grow[k];

        float hn16[H_DIM];                   // 16*h_new, integers in [-16,16]
#pragma unroll
        for (int k = 0; k < H_DIM; ++k) {
            const float pi = rintf(g[k]);
            const float pf = rintf(g[H_DIM + k]);
            const float pg = rintf(g[2 * H_DIM + k]);
            const float po = rintf(g[3 * H_DIM + k]);
            const float ia = qhsig16(pi);                      // [0,16]
            const float fa = qhsig16(pf);                      // [0,16]
            const float oa = qhsig16(po);                      // [0,16]
            const float ga = fminf(fmaxf(pg, -16.0f), 16.0f);  // 16*tanh-hat
            // S = 256*(f*c + i*g); 16*c_new = clamp(rint(S/16), -128, 127)
            const float S  = __builtin_fmaf(fa, c16[k], ia * ga);
            const float cn = fminf(fmaxf(rintf(S * 0.0625f), -128.0f), 127.0f);
            c16[k] = cn;
            const float th = fminf(fmaxf(cn, -16.0f), 16.0f);  // 16*q(htanh(c))
            hn16[k] = rintf(oa * th * 0.0625f);                // 16*h_new
        }

        // -- output head: y = clamp(rint(16*(h.Wl + bl)), -128, 127)/16 -----
        float acc16 = bl16;
#pragma unroll
        for (int k = 0; k < H_DIM; ++k) acc16 = __builtin_fmaf(hn16[k], wl[k], acc16);
        yb[t] = fminf(fmaxf(rintf(acc16), -128.0f), 127.0f) * 0.0625f;

        // -- pack next A row: [xq(t+1) | 16*h(t)] ---------------------------
        float nx0 = 0.0f, nx1 = 0.0f, nx2 = 0.0f;
        if (t + 1 < T_LEN) {                 // uniform branch
            const float* xp = xb + (size_t)(t + 1) * I_DIM;
            nx0 = q4(xp[0]); nx1 = q4(xp[1]); nx2 = q4(xp[2]);
        }
        ARow nr;
        nr.h[0] = (_Float16)nx0;
        nr.h[1] = (_Float16)nx1;
        nr.h[2] = (_Float16)nx2;
#pragma unroll
        for (int k = 0; k < H_DIM; ++k) nr.h[I_DIM + k] = (_Float16)hn16[k];
        *(ARow*)&hp[lane] = nr;              // upper 16 B zeros stay intact
    }
}

extern "C" void kernel_launch(void* const* d_in, const int* in_sizes, int n_in,
                              void* d_out, int out_size, void* d_ws, size_t ws_size,
                              hipStream_t stream) {
    (void)in_sizes; (void)n_in; (void)out_size; (void)d_ws; (void)ws_size;
    const float* x     = (const float*)d_in[0];
    const float* W_ih  = (const float*)d_in[1];
    const float* W_hh  = (const float*)d_in[2];
    const float* b_ih  = (const float*)d_in[3];
    const float* b_hh  = (const float*)d_in[4];
    const float* W_lin = (const float*)d_in[5];
    const float* b_lin = (const float*)d_in[6];
    float* yo = (float*)d_out;

    // 4096 batch rows, 32 per wave, one wave32 per workgroup -> 128 blocks.
    lstm_qfixed_wmma<<<B_TOT / 32, 32, 0, stream>>>(x, W_ih, W_hh, b_ih, b_hh,
                                                    W_lin, b_lin, yo);
}
